// CrossModalFusionBlockSE_43319040147655
// MI455X (gfx1250) — matrix operand order repaired and checked by reading, not compiled, not run on hardware
//
#include <hip/hip_runtime.h>
#include <hip/hip_bf16.h>

typedef __attribute__((ext_vector_type(16))) _Float16 v16h;
typedef __attribute__((ext_vector_type(4)))  _Float16 v4h;
typedef __attribute__((ext_vector_type(8)))  float    v8f;

constexpr int BATCH = 65536;
constexpr int STR   = 200;                  // padded LDS row stride (floats), avoids 64-bank conflicts
constexpr int WAVES = 4;                    // waves per block (wave32)
constexpr int WAVE_F = 3 * 16 * STR + 192;  // tokb + b1 + b2 + softmax buf (floats per wave)

// ---- f16 weight workspace layout (element offsets; all 32B aligned) ----
constexpr size_t OFF_PW0 = 0;                       // 64x512
constexpr size_t OFF_PW1 = OFF_PW0 + 64 * 512;      // 64x256
constexpr size_t OFF_PW2 = OFF_PW1 + 64 * 256;      // 64x128
constexpr size_t OFF_RW0 = OFF_PW2 + 64 * 128;      // 512x64
constexpr size_t OFF_RW1 = OFF_RW0 + 512 * 64;      // 256x64
constexpr size_t OFF_RW2 = OFF_RW1 + 256 * 64;      // 128x64
constexpr size_t OFF_QKV = OFF_RW2 + 128 * 64;      // 192x64
constexpr size_t OFF_OUT = OFF_QKV + 192 * 64;      // 64x64
constexpr size_t OFF_PWW = OFF_OUT + 64 * 64;       // 3x64x128
constexpr size_t OFF_SE1 = OFF_PWW + 3 * 64 * 128;  // 3x32x64
constexpr size_t OFF_SE2 = OFF_SE1 + 3 * 32 * 64;   // 3x64x32
constexpr size_t OFF_FCW = OFF_SE2 + 3 * 64 * 32;   // 3x64x64
constexpr size_t WS_ELEMS = OFF_FCW + 3 * 64 * 64;  // 180224 f16 = 352 KB

// ---------------- prologue: pack f32 weights -> f16 workspace ----------------
extern "C" __global__ void __launch_bounds__(256, 1)
pack_w_f16(const float* __restrict__ src, _Float16* __restrict__ dst, int n4) {
    int i = blockIdx.x * blockDim.x + threadIdx.x;
    if (i < n4) {
        float4 v = ((const float4*)src)[i];
        v4h o;
        o[0] = (_Float16)v.x; o[1] = (_Float16)v.y;
        o[2] = (_Float16)v.z; o[3] = (_Float16)v.w;
        ((v4h*)dst)[i] = o;
    }
}

// ---------------- WMMA wrapper ----------------
__device__ __forceinline__ v8f wmma16(v16h a, v16h b, v8f c) {
    // D = A(16x32 f16) x B(32x16 f16) + C(16x16 f32)
    return __builtin_amdgcn_wmma_f32_16x16x32_f16(false, a, false, b, (short)0, c, false, false);
}

__device__ __forceinline__ v8f splat8(float x) {
    v8f r;
#pragma unroll
    for (int i = 0; i < 8; ++i) r[i] = x;
    return r;
}

// A-layout 16x32 f16 tile from LDS f32 (p = base + row*STR + colOff).
// Lane: row = lane&15; kh = lane>>4.  elems 0..7 -> K = kt+kh*8 .. ; 8..15 -> K = kt+16+kh*8 ..
__device__ __forceinline__ v16h ldA_lds(const float* p, int kt, int kh) {
    v16h a;
    const int k0 = kt + kh * 8, k1 = kt + 16 + kh * 8;
#pragma unroll
    for (int j = 0; j < 8; ++j) {
        a[j]     = (_Float16)p[k0 + j];
        a[j + 8] = (_Float16)p[k1 + j];
    }
    return a;
}

// Same, with fused depthwise k=1 conv (per-channel scale+bias), channel index == K index.
__device__ __forceinline__ v16h ldA_dw(const float* p, int kt, int kh,
                                       const float* sw, const float* sb) {
    v16h a;
    const int k0 = kt + kh * 8, k1 = kt + 16 + kh * 8;
#pragma unroll
    for (int j = 0; j < 8; ++j) {
        a[j]     = (_Float16)(p[k0 + j] * sw[k0 + j] + sb[k0 + j]);
        a[j + 8] = (_Float16)(p[k1 + j] * sw[k1 + j] + sb[k1 + j]);
    }
    return a;
}

// A tile straight from global f32 (p = rowbase + kt), vectorized 128-bit loads.
__device__ __forceinline__ v16h ldA_glb(const float* p, int kh) {
    const float4* q0 = (const float4*)(p + kh * 8);
    const float4* q1 = (const float4*)(p + 16 + kh * 8);
    float4 a0 = q0[0], a1 = q0[1], b0 = q1[0], b1 = q1[1];
    v16h a;
    a[0]  = (_Float16)a0.x; a[1]  = (_Float16)a0.y; a[2]  = (_Float16)a0.z; a[3]  = (_Float16)a0.w;
    a[4]  = (_Float16)a1.x; a[5]  = (_Float16)a1.y; a[6]  = (_Float16)a1.z; a[7]  = (_Float16)a1.w;
    a[8]  = (_Float16)b0.x; a[9]  = (_Float16)b0.y; a[10] = (_Float16)b0.z; a[11] = (_Float16)b0.w;
    a[12] = (_Float16)b1.x; a[13] = (_Float16)b1.y; a[14] = (_Float16)b1.z; a[15] = (_Float16)b1.w;
    return a;
}

// B-layout 32x16 f16 tile from the pre-packed f16 workspace: B[k][n] = W[nbase+n][kt+k].
// Lane: n = lane&15; kh = lane>>4: elems 0..7 -> K = kt+kh*8 .., 8..15 -> K = kt+16+kh*8 .. (two 16B loads, no cvt).
__device__ __forceinline__ v16h ldB_h(const _Float16* __restrict__ W, int ld,
                                      int nbase, int kt, int col, int kh) {
    typedef __attribute__((ext_vector_type(8))) _Float16 v8h_;
    const _Float16* p = W + (size_t)(nbase + col) * ld + kt + kh * 8;
    v8h_ lo = *(const v8h_*)p;
    v8h_ hi = *(const v8h_*)(p + 16);
    return __builtin_shufflevector(lo, hi, 0, 1, 2, 3, 4, 5, 6, 7, 8, 9, 10, 11, 12, 13, 14, 15);
}

// Store a 16x16 f32 C/D tile to LDS (dst = base + colOff). Row = v + 8*kh, col = lane&15.
__device__ __forceinline__ void stT(float* dst, v8f acc, int col, int kh) {
#pragma unroll
    for (int v = 0; v < 8; ++v) dst[(size_t)(v + 8 * kh) * STR + col] = acc[v];
}

extern "C" __global__ void __launch_bounds__(128, 1)
xmodal_fused_kernel(const float* __restrict__ f0, const float* __restrict__ f1,
                    const float* __restrict__ f2,
                    const float* __restrict__ pb0, const float* __restrict__ pb1,
                    const float* __restrict__ pb2,
                    const float* __restrict__ rb0, const float* __restrict__ rb1,
                    const float* __restrict__ rb2,
                    const float* __restrict__ qkvb, const float* __restrict__ outb,
                    const float* __restrict__ dww,  const float* __restrict__ dwb,
                    const float* __restrict__ pwb,
                    const float* __restrict__ se1b, const float* __restrict__ se2b,
                    const float* __restrict__ fcb,
                    const _Float16* __restrict__ wsh,   // packed f16 weights
                    float* __restrict__ out) {
    extern __shared__ float smem[];
    const int lane = threadIdx.x & 31;
    const int wave = threadIdx.x >> 5;
    const int col  = lane & 15;   // also the batch row this lane owns for per-row math
    const int kh   = lane >> 4;
    const int r0   = (blockIdx.x * WAVES + wave) * 16;

    float* tokb = smem + (size_t)wave * WAVE_F;  // tokens   (16 x 192)
    float* b1   = tokb + 16 * STR;               // scratch  (16 x 192): Q / V / attn-in / gate tmp / new
    float* b2   = b1 + 16 * STR;                 // scratch  (16 x 192): K / attn-out
    float* ab   = b2 + 16 * STR;                 // softmax weights (16 x 9, padded 12)

    const float* fs[3]  = {f0, f1, f2};
    const float* pbs[3] = {pb0, pb1, pb2};
    const float* rbs[3] = {rb0, rb1, rb2};
    const int    dsz[3] = {512, 256, 128};
    const _Float16* pws[3] = {wsh + OFF_PW0, wsh + OFF_PW1, wsh + OFF_PW2};
    const _Float16* rws[3] = {wsh + OFF_RW0, wsh + OFF_RW1, wsh + OFF_RW2};
    const _Float16* qkvw = wsh + OFF_QKV;
    const _Float16* outw = wsh + OFF_OUT;
    float* outs[3] = {out,
                      out + (size_t)BATCH * 512,
                      out + (size_t)BATCH * 512 + (size_t)BATCH * 256};

    // ================= Phase 1: projections  tokens[b, m*64+c] =================
#pragma unroll
    for (int m = 0; m < 3; ++m) {
        const int d = dsz[m];
        const float* frow = fs[m] + (size_t)(r0 + col) * d;  // this lane's batch row
        __builtin_prefetch(frow + (size_t)64 * d, 0, 1);     // next block's tile (L2 warm)
        v8f acc[4];
#pragma unroll
        for (int n = 0; n < 4; ++n) acc[n] = splat8(pbs[m][n * 16 + col]);
        for (int kt = 0; kt < d; kt += 32) {
            v16h a = ldA_glb(frow + kt, kh);
#pragma unroll
            for (int n = 0; n < 4; ++n)
                acc[n] = wmma16(a, ldB_h(pws[m], d, n * 16, kt, col, kh), acc[n]);
        }
#pragma unroll
        for (int n = 0; n < 4; ++n) stT(tokb + m * 64 + n * 16, acc[n], col, kh);
    }

    // ================= Phase 2a: Q -> b1, K -> b2 =================
#pragma unroll
    for (int m = 0; m < 3; ++m) {
        const float* tp = tokb + (size_t)col * STR + m * 64;
        v16h a0 = ldA_lds(tp, 0, kh);
        v16h a1 = ldA_lds(tp, 32, kh);
#pragma unroll
        for (int half = 0; half < 2; ++half) {
            float* dst = (half == 0) ? b1 : b2;
            const int eb = half * 64;
            v8f acc[4];
#pragma unroll
            for (int n = 0; n < 4; ++n) acc[n] = splat8(qkvb[eb + n * 16 + col]);
#pragma unroll
            for (int n = 0; n < 4; ++n) {
                acc[n] = wmma16(a0, ldB_h(qkvw, 64, eb + n * 16, 0, col, kh), acc[n]);
                acc[n] = wmma16(a1, ldB_h(qkvw, 64, eb + n * 16, 32, col, kh), acc[n]);
            }
#pragma unroll
            for (int n = 0; n < 4; ++n) stT(dst + m * 64 + n * 16, acc[n], col, kh);
        }
    }

    // ================= Phase 2b: 3x3 scores + softmax (per row, split across lane halves) ======
    {
        float a9[9];
        const int cb = kh * 32;
#pragma unroll
        for (int mq = 0; mq < 3; ++mq) {
#pragma unroll
            for (int mk = 0; mk < 3; ++mk) {
                float s = 0.f;
                const float* qp = b1 + (size_t)col * STR + mq * 64 + cb;
                const float* kp = b2 + (size_t)col * STR + mk * 64 + cb;
#pragma unroll
                for (int c = 0; c < 32; c += 4) {
                    float4 qv = *(const float4*)(qp + c);
                    float4 kv = *(const float4*)(kp + c);
                    s += qv.x * kv.x + qv.y * kv.y + qv.z * kv.z + qv.w * kv.w;
                }
                s += __shfl_xor(s, 16);          // combine the two 32-channel halves
                a9[mq * 3 + mk] = s * 0.125f;    // 1/sqrt(64)
            }
        }
#pragma unroll
        for (int mq = 0; mq < 3; ++mq) {
            float e0 = a9[mq * 3], e1 = a9[mq * 3 + 1], e2 = a9[mq * 3 + 2];
            float mx = fmaxf(e0, fmaxf(e1, e2));
            float x0 = __expf(e0 - mx), x1 = __expf(e1 - mx), x2 = __expf(e2 - mx);
            float inv = 1.f / (x0 + x1 + x2);
            a9[mq * 3] = x0 * inv; a9[mq * 3 + 1] = x1 * inv; a9[mq * 3 + 2] = x2 * inv;
        }
        if (kh == 0) {
#pragma unroll
            for (int i = 0; i < 9; ++i) ab[col * 12 + i] = a9[i];
        }
    }

    // ================= Phase 2c: V -> b1 (overwrite Q, scores already consumed) =================
#pragma unroll
    for (int m = 0; m < 3; ++m) {
        const float* tp = tokb + (size_t)col * STR + m * 64;
        v16h a0 = ldA_lds(tp, 0, kh);
        v16h a1 = ldA_lds(tp, 32, kh);
        v8f acc[4];
#pragma unroll
        for (int n = 0; n < 4; ++n) acc[n] = splat8(qkvb[128 + n * 16 + col]);
#pragma unroll
        for (int n = 0; n < 4; ++n) {
            acc[n] = wmma16(a0, ldB_h(qkvw, 64, 128 + n * 16, 0, col, kh), acc[n]);
            acc[n] = wmma16(a1, ldB_h(qkvw, 64, 128 + n * 16, 32, col, kh), acc[n]);
        }
#pragma unroll
        for (int n = 0; n < 4; ++n) stT(b1 + m * 64 + n * 16, acc[n], col, kh);
    }

    // ================= Phase 2d: attn = softmax @ V, in-place in b1 =================
    {
        float aw[9];
#pragma unroll
        for (int i = 0; i < 9; ++i) aw[i] = ab[col * 12 + i];
        float* vp = b1 + (size_t)col * STR + kh * 32;
#pragma unroll
        for (int c = 0; c < 32; c += 4) {
            float4 v0 = *(float4*)(vp + 0 * 64 + c);
            float4 v1 = *(float4*)(vp + 1 * 64 + c);
            float4 v2 = *(float4*)(vp + 2 * 64 + c);
#pragma unroll
            for (int m = 0; m < 3; ++m) {
                float4 o;
                o.x = aw[m * 3] * v0.x + aw[m * 3 + 1] * v1.x + aw[m * 3 + 2] * v2.x;
                o.y = aw[m * 3] * v0.y + aw[m * 3 + 1] * v1.y + aw[m * 3 + 2] * v2.y;
                o.z = aw[m * 3] * v0.z + aw[m * 3 + 1] * v1.z + aw[m * 3 + 2] * v2.z;
                o.w = aw[m * 3] * v0.w + aw[m * 3 + 1] * v1.w + aw[m * 3 + 2] * v2.w;
                *(float4*)(vp + m * 64 + c) = o;
            }
        }
    }

    // ================= Phase 2e: out-proj -> b2 (overwrite K) =================
#pragma unroll
    for (int m = 0; m < 3; ++m) {
        const float* apn = b1 + (size_t)col * STR + m * 64;
        v16h a0 = ldA_lds(apn, 0, kh);
        v16h a1 = ldA_lds(apn, 32, kh);
        v8f acc[4];
#pragma unroll
        for (int n = 0; n < 4; ++n) acc[n] = splat8(outb[n * 16 + col]);
#pragma unroll
        for (int n = 0; n < 4; ++n) {
            acc[n] = wmma16(a0, ldB_h(outw, 64, n * 16, 0, col, kh), acc[n]);
            acc[n] = wmma16(a1, ldB_h(outw, 64, n * 16, 32, col, kh), acc[n]);
        }
#pragma unroll
        for (int n = 0; n < 4; ++n) stT(b2 + m * 64 + n * 16, acc[n], col, kh);
    }

    // ================= Phase 3: gates + residual + reverse projection, per modality =============
#pragma unroll
    for (int m = 0; m < 3; ++m) {
        const float*    dwwm = dww + m * 128;           const float* dwbm = dwb + m * 128;
        const _Float16* pwm  = wsh + OFF_PWW + m * 64 * 128;
        const _Float16* s1wm = wsh + OFF_SE1 + m * 32 * 64;
        const _Float16* s2wm = wsh + OFF_SE2 + m * 64 * 32;
        const _Float16* fcwm = wsh + OFF_FCW + m * 64 * 64;
        const float* pwbm = pwb + m * 64;
        const float* s1bm = se1b + m * 32;
        const float* s2bm = se2b + m * 64;
        const float* fcbm = fcb + m * 64;

        // pointwise conv: x = dw(gate_in) folded into A-load; K 0..63 from tokens, 64..127 from attn
        const float* tp  = tokb + (size_t)col * STR + m * 64;
        const float* apb = b2 + (size_t)col * STR + (m * 64 - 64);  // deref only at k>=64
        v16h x0 = ldA_dw(tp, 0, kh, dwwm, dwbm);
        v16h x1 = ldA_dw(tp, 32, kh, dwwm, dwbm);
        v16h x2 = ldA_dw(apb, 64, kh, dwwm, dwbm);
        v16h x3 = ldA_dw(apb, 96, kh, dwwm, dwbm);

        v8f hacc[4];
#pragma unroll
        for (int n = 0; n < 4; ++n) hacc[n] = splat8(pwbm[n * 16 + col]);
#pragma unroll
        for (int n = 0; n < 4; ++n) {
            hacc[n] = wmma16(x0, ldB_h(pwm, 128, n * 16, 0, col, kh), hacc[n]);
            hacc[n] = wmma16(x1, ldB_h(pwm, 128, n * 16, 32, col, kh), hacc[n]);
            hacc[n] = wmma16(x2, ldB_h(pwm, 128, n * 16, 64, col, kh), hacc[n]);
            hacc[n] = wmma16(x3, ldB_h(pwm, 128, n * 16, 96, col, kh), hacc[n]);
        }
#pragma unroll
        for (int n = 0; n < 4; ++n) {
#pragma unroll
            for (int v = 0; v < 8; ++v) hacc[n][v] = fmaxf(hacc[n][v], 0.f);
            stT(b1 + n * 16, hacc[n], col, kh);  // h -> b1 cols 0..63
        }

        // SE fc1 (64 -> 32) + relu -> b1 cols 64..95
        const float* hp = b1 + (size_t)col * STR;
        {
            v16h h0 = ldA_lds(hp, 0, kh);
            v16h h1 = ldA_lds(hp, 32, kh);
#pragma unroll
            for (int n = 0; n < 2; ++n) {
                v8f s = splat8(s1bm[n * 16 + col]);
                s = wmma16(h0, ldB_h(s1wm, 64, n * 16, 0, col, kh), s);
                s = wmma16(h1, ldB_h(s1wm, 64, n * 16, 32, col, kh), s);
#pragma unroll
                for (int v = 0; v < 8; ++v) s[v] = fmaxf(s[v], 0.f);
                stT(b1 + 64 + n * 16, s, col, kh);
            }
        }

        // SE fc2 (32 -> 64), sigmoid, h *= s ; hs -> b1 cols 0..63
        {
            v16h s0 = ldA_lds(hp + 64, 0, kh);  // single 16x32 A tile
#pragma unroll
            for (int n = 0; n < 4; ++n) {
                v8f g = splat8(s2bm[n * 16 + col]);
                g = wmma16(s0, ldB_h(s2wm, 32, n * 16, 0, col, kh), g);
#pragma unroll
                for (int v = 0; v < 8; ++v)
                    hacc[n][v] *= 1.f / (1.f + __expf(-g[v]));
                stT(b1 + n * 16, hacc[n], col, kh);
            }
        }

        // final gate FC (64 -> 64)
        v8f gt[4];
        {
            v16h hs0 = ldA_lds(hp, 0, kh);
            v16h hs1 = ldA_lds(hp, 32, kh);
#pragma unroll
            for (int n = 0; n < 4; ++n) {
                gt[n] = splat8(fcbm[n * 16 + col]);
                gt[n] = wmma16(hs0, ldB_h(fcwm, 64, n * 16, 0, col, kh), gt[n]);
                gt[n] = wmma16(hs1, ldB_h(fcwm, 64, n * 16, 32, col, kh), gt[n]);
            }
        }

        // new = tokens + gate * attn  -> b1 cols 0..63 (elementwise in C/D layout)
#pragma unroll
        for (int n = 0; n < 4; ++n) {
#pragma unroll
            for (int v = 0; v < 8; ++v) {
                const size_t row = (size_t)(v + 8 * kh);
                float t  = tokb[row * STR + m * 64 + n * 16 + col];
                float at = b2[row * STR + m * 64 + n * 16 + col];
                b1[row * STR + n * 16 + col] = t + gt[n][v] * at;
            }
        }

        // reverse projection: (16 x 64) @ (64 x d) + rev_b -> global out
        {
            v16h n0 = ldA_lds(hp, 0, kh);
            v16h n1 = ldA_lds(hp, 32, kh);
            const int d = dsz[m];
            const _Float16* rw = rws[m];
            const float* rb = rbs[m];
            float* outm = outs[m] + (size_t)r0 * d;
            for (int nt = 0; nt < d / 16; ++nt) {
                v8f r = splat8(rb[nt * 16 + col]);
                r = wmma16(n0, ldB_h(rw, 64, nt * 16, 0, col, kh), r);
                r = wmma16(n1, ldB_h(rw, 64, nt * 16, 32, col, kh), r);
#pragma unroll
                for (int v = 0; v < 8; ++v)
                    outm[(size_t)(v + 8 * kh) * d + nt * 16 + col] = r[v];
            }
        }
    }
}

extern "C" void kernel_launch(void* const* d_in, const int* in_sizes, int n_in,
                              void* d_out, int out_size, void* d_ws, size_t ws_size,
                              hipStream_t stream) {
    (void)in_sizes; (void)n_in; (void)out_size; (void)ws_size;
    const float* f0   = (const float*)d_in[0];
    const float* pw0  = (const float*)d_in[1];
    const float* pb0  = (const float*)d_in[2];
    const float* rw0  = (const float*)d_in[3];
    const float* rb0  = (const float*)d_in[4];
    const float* f1   = (const float*)d_in[5];
    const float* pw1  = (const float*)d_in[6];
    const float* pb1  = (const float*)d_in[7];
    const float* rw1  = (const float*)d_in[8];
    const float* rb1  = (const float*)d_in[9];
    const float* f2   = (const float*)d_in[10];
    const float* pw2  = (const float*)d_in[11];
    const float* pb2  = (const float*)d_in[12];
    const float* rw2  = (const float*)d_in[13];
    const float* rb2  = (const float*)d_in[14];
    const float* qkvw = (const float*)d_in[15];
    const float* qkvb = (const float*)d_in[16];
    const float* outw = (const float*)d_in[17];
    const float* outb = (const float*)d_in[18];
    const float* dww  = (const float*)d_in[19];
    const float* dwb  = (const float*)d_in[20];
    const float* pww  = (const float*)d_in[21];
    const float* pwb  = (const float*)d_in[22];
    const float* s1w  = (const float*)d_in[23];
    const float* s1b  = (const float*)d_in[24];
    const float* s2w  = (const float*)d_in[25];
    const float* s2b  = (const float*)d_in[26];
    const float* fcw  = (const float*)d_in[27];
    const float* fcb  = (const float*)d_in[28];

    _Float16* wsh = (_Float16*)d_ws;  // 352 KB of packed f16 weights

    // ---- prologue: convert all weight matrices to f16 in the workspace ----
    struct { const float* s; size_t off; int n; } packs[12] = {
        {pw0,  OFF_PW0, 64 * 512}, {pw1, OFF_PW1, 64 * 256}, {pw2, OFF_PW2, 64 * 128},
        {rw0,  OFF_RW0, 512 * 64}, {rw1, OFF_RW1, 256 * 64}, {rw2, OFF_RW2, 128 * 64},
        {qkvw, OFF_QKV, 192 * 64}, {outw, OFF_OUT, 64 * 64},
        {pww,  OFF_PWW, 3 * 64 * 128}, {s1w, OFF_SE1, 3 * 32 * 64},
        {s2w,  OFF_SE2, 3 * 64 * 32},  {fcw, OFF_FCW, 3 * 64 * 64},
    };
    for (int i = 0; i < 12; ++i) {
        int n4 = packs[i].n / 4;
        pack_w_f16<<<(n4 + 255) / 256, 256, 0, stream>>>(packs[i].s, wsh + packs[i].off, n4);
    }

    // ---- fused main kernel ----
    dim3 grid(BATCH / (WAVES * 16));   // 1024 workgroups, each = 4 waves x 16 rows
    dim3 block(WAVES * 32);            // 128 threads (wave32)
    size_t shmem = (size_t)WAVES * WAVE_F * sizeof(float);  // ~153 KB of the 320 KB WGP LDS

    xmodal_fused_kernel<<<grid, block, shmem, stream>>>(
        f0, f1, f2,
        pb0, pb1, pb2,
        rb0, rb1, rb2,
        qkvb, outb,
        dww, dwb, pwb,
        s1b, s2b, fcb,
        wsh,
        (float*)d_out);
}
